// LIF_SNN_88441966559366
// MI455X (gfx1250) — matrix operand order repaired and checked by reading, not compiled, not run on hardware
//
#include <hip/hip_runtime.h>

// MI455X / gfx1250 persistent SNN scan.
//   - grid: 8 workgroups (one 16-row batch tile each), 1024 threads = 32 wave32
//   - recurrence is batch-row-local -> no cross-WG sync needed, the full
//     T=256 scan runs in one kernel with 2 s_barriers per step.
//   - matmul s@Wrec^T done with v_wmma_f32_16x16x32_f16 (spikes are exactly
//     representable in f16); Wrec pre-packed to f16 B-fragment layout in d_ws.
//   - spike state bit-packed in one VGPR (16 bits) to avoid register spills.

typedef __attribute__((ext_vector_type(16))) _Float16 v16h;
typedef __attribute__((ext_vector_type(8)))  float    v8f;

#define B_  128
#define T_  256
#define N_  1024
#define KPAD 8          // 16B row pad -> conflict-free A-fragment ds_load_b128

union Frag { v16h h; float4 q[2]; };

// Pack Wrec (N x N f32, row-major, rec[b,n] = sum_k s[b,k]*Wrec[n,k]) into
// f16 B-fragment layout (nt=0..63): at k-chunk kc, lane l of a wave reads 16
// contiguous f16 at bpack[(nt*N + kc*32 + l)*16] == col n0 + l%16, K = kc*32 +
// 8*(l/16) + s (s<8) / 16 + 8*(l/16) + s-8 (s>=8).
__global__ __launch_bounds__(1024)
void pack_wrec_f16(const float* __restrict__ wrec, _Float16* __restrict__ bpack) {
    const int n = blockIdx.x;          // 0..1023  (output neuron)
    const int k = threadIdx.x;         // 0..1023  (K index)
    bpack[((size_t)(n >> 4) * N_ + (k & ~31) + (n & 15) + (k & 8) * 2) * 16 + (k & 7) + ((k >> 1) & 8)] =
        (_Float16)wrec[(size_t)n * N_ + k];
}

__global__ __launch_bounds__(1024, 1)
void snn_scan(const float* __restrict__ xin,    // (B,T)   input_batch
              const float* __restrict__ st0,    // (3,B,N) init state
              const float* __restrict__ li0,    // (B,)    init LI
              const float* __restrict__ w1,     // (N,)    l1_weights
              const float* __restrict__ bias1,  // (N,)
              const float* __restrict__ thres,  // (N,)
              const float* __restrict__ lki,    // (N,)
              const float* __restrict__ lkv,    // (N,)
              const float* __restrict__ l2w,    // (N,)
              const float* __restrict__ l2leak, // (1,)
              const _Float16* __restrict__ bpack,
              float* __restrict__ out_spk,      // (T,B,N)
              float* __restrict__ out_st,       // (T,3,B,N)
              float* __restrict__ out_dec)      // (T,B)
{
    __shared__ _Float16 sbuf[16][N_ + KPAD];    // spikes of the 16 batch rows, f16
    __shared__ float    liacc[16];

    const int bbase = blockIdx.x * 16;          // batch tile
    const int wave  = threadIdx.x >> 5;         // 0..31 -> 32-neuron slab
    const int lane  = threadIdx.x & 31;
    const int col   = lane & 15;                // D-tile column
    const int lh    = lane >> 4;                // lane half (row group)
    const int nbase = wave * 32;

    // ---- per-neuron params in D-tile layout (2 tiles of 16 cols) ----
    float pLkI[2], pLkV[2], pBia[2], pThr[2], pW1[2], pL2[2];
    #pragma unroll
    for (int tl = 0; tl < 2; ++tl) {
        const int n = nbase + tl * 16 + col;
        pLkI[tl] = lki[n];   pLkV[tl] = lkv[n];
        pBia[tl] = bias1[n]; pThr[tl] = thres[n];
        pW1[tl]  = w1[n];    pL2[tl]  = l2w[n];
    }
    const float l2lk = l2leak[0];

    // ---- initial state: i,v in regs; spike state bit-packed in sMask ----
    float iS[2][8], vS[2][8];
    unsigned int sMask = 0;                     // bit (tl*8 + r) = spike(m,n)
    #pragma unroll
    for (int tl = 0; tl < 2; ++tl) {
        const int n = nbase + tl * 16 + col;
        #pragma unroll
        for (int r = 0; r < 8; ++r) {
            const int m = r + 8 * lh;
            const size_t off = (size_t)(bbase + m) * N_ + n;
            iS[tl][r] = st0[0 * (size_t)B_ * N_ + off];
            vS[tl][r] = st0[1 * (size_t)B_ * N_ + off];
            const float s0 = st0[2 * (size_t)B_ * N_ + off];
            if (s0 != 0.0f) sMask |= (1u << (tl * 8 + r));
            sbuf[m][n] = (_Float16)s0;
        }
    }
    float liReg = 0.0f;
    if (wave == 0 && lane < 16) { liReg = li0[bbase + lane]; liacc[lane] = 0.0f; }
    __syncthreads();

    const int mA   = lane & 15;      // A-fragment row this lane supplies
    const int offA = lh * 8;         // K sub-offset per documented A layout

    for (int t = 0; t < T_; ++t) {
        // -------- rec = S(16xK) @ WrecT : 64 chained WMMAs per wave --------
        v8f acc0 = {}; v8f acc1 = {};
        #pragma unroll 2
        for (int kc = 0; kc < N_ / 32; ++kc) {
            const int k0 = kc * 32;
            Frag a;      // A 16x32 f16: v0-3 = K offA..offA+7, v4-7 = K+16
            a.q[0] = *(const float4*)&sbuf[mA][k0 + offA];
            a.q[1] = *(const float4*)&sbuf[mA][k0 + 16 + offA];
            const _Float16* bp0 = bpack + ((size_t)(nbase >> 4) * N_ + k0 + lane) * 16;
            const _Float16* bp1 = bp0 + (size_t)N_ * 16;
            Frag b0, b1;
            b0.q[0] = ((const float4*)bp0)[0]; b0.q[1] = ((const float4*)bp0)[1];
            b1.q[0] = ((const float4*)bp1)[0]; b1.q[1] = ((const float4*)bp1)[1];
            acc0 = __builtin_amdgcn_wmma_f32_16x16x32_f16(false, a.h, false, b0.h,
                                                          (short)0, acc0, false, false);
            acc1 = __builtin_amdgcn_wmma_f32_16x16x32_f16(false, a.h, false, b1.h,
                                                          (short)0, acc1, false, false);
        }
        __syncthreads();   // all sbuf reads of step t done -> safe to overwrite

        // -------- elementwise LIF update + output stores --------
        float xrow[8];
        #pragma unroll
        for (int r = 0; r < 8; ++r)
            xrow[r] = xin[(size_t)(bbase + r + 8 * lh) * T_ + t];

        const size_t spkBase = (size_t)t * B_ * N_;
        const size_t stBase  = (size_t)t * 3 * B_ * N_;
        unsigned int newMask = 0;
        #pragma unroll
        for (int tl = 0; tl < 2; ++tl) {
            const int n = nbase + tl * 16 + col;
            float* __restrict__ stp = out_st + stBase + (size_t)bbase * N_ + n;
            #pragma unroll
            for (int r = 0; r < 8; ++r) {
                const int m   = r + 8 * lh;
                const int bit = tl * 8 + r;
                const float rec  = (tl == 0) ? acc0[r] : acc1[r];
                const float inew = pLkI[tl] * iS[tl][r] + xrow[r] * pW1[tl]
                                 + pBia[tl] + rec;
                // hard reset: v = lkv*v*(1-s_prev) + i  ->  cndmask on spike bit
                const float keep = (sMask & (1u << bit)) ? 0.0f : 1.0f;
                const float vnew = pLkV[tl] * vS[tl][r] * keep + inew;
                const bool  spk  = (vnew - pThr[tl] > 0.0f);
                const float snew = spk ? 1.0f : 0.0f;
                if (spk) newMask |= (1u << bit);
                iS[tl][r] = inew; vS[tl][r] = vnew;
                sbuf[m][n] = (_Float16)snew;                 // feeds step t+1
                out_spk[spkBase + (size_t)(bbase + m) * N_ + n] = snew;
                float* p = stp + (size_t)m * N_;             // 3 slabs, imm offsets
                p[0 * (size_t)B_ * N_] = inew;
                p[1 * (size_t)B_ * N_] = vnew;
                p[2 * (size_t)B_ * N_] = snew;
            }
        }
        sMask = newMask;

        // -------- LI readout: per-row reduction across lanes+waves --------
        #pragma unroll
        for (int r = 0; r < 8; ++r) {
            float p = ((newMask >> r)       & 1u ? pL2[0] : 0.0f)
                    + ((newMask >> (8 + r)) & 1u ? pL2[1] : 0.0f);
            p += __shfl_xor(p, 1, 32);
            p += __shfl_xor(p, 2, 32);
            p += __shfl_xor(p, 4, 32);
            p += __shfl_xor(p, 8, 32);   // halves stay separate (masks < 16)
            if (col == 0) atomicAdd(&liacc[r + 8 * lh], p);  // ds_add_f32
        }
        __syncthreads();   // liacc complete; sbuf(t+1) visible
        if (wave == 0 && lane < 16) {
            liReg = l2lk * liReg + liacc[lane];
            out_dec[(size_t)t * B_ + bbase + lane] = liReg;
            liacc[lane] = 0.0f;          // protected by next step's barrier
        }
    }
}

extern "C" void kernel_launch(void* const* d_in, const int* in_sizes, int n_in,
                              void* d_out, int out_size, void* d_ws, size_t ws_size,
                              hipStream_t stream) {
    const float* xin   = (const float*)d_in[0];   // (B,T,1)
    const float* st0   = (const float*)d_in[1];   // (3,B,N)
    const float* li0   = (const float*)d_in[2];   // (B,1)
    const float* w1    = (const float*)d_in[3];   // (N,1)
    const float* b1    = (const float*)d_in[4];   // (N,)
    const float* wrec  = (const float*)d_in[5];   // (N,N)
    const float* thres = (const float*)d_in[6];
    const float* lkiP  = (const float*)d_in[7];
    const float* lkvP  = (const float*)d_in[8];
    const float* l2w   = (const float*)d_in[9];   // (1,N)
    const float* l2lk  = (const float*)d_in[10];  // (1,)

    float* out      = (float*)d_out;
    float* out_spk  = out;                                  // (T,B,N)
    float* out_st   = out + (size_t)T_ * B_ * N_;           // (T,3,B,N)
    float* out_dec  = out + (size_t)T_ * B_ * N_ * 4;       // (T,B)
    _Float16* bpack = (_Float16*)d_ws;                      // 2 MB f16 Wrec pack

    pack_wrec_f16<<<N_, 1024, 0, stream>>>(wrec, bpack);
    snn_scan<<<B_ / 16, 1024, 0, stream>>>(xin, st0, li0, w1, b1, thres,
                                           lkiP, lkvP, l2w, l2lk, bpack,
                                           out_spk, out_st, out_dec);
}
